// PtsCrossAttention_PyTorch_34291018891936
// MI455X (gfx1250) — matrix operand order repaired and checked by reading, not compiled, not run on hardware
//
#include <hip/hip_runtime.h>
#include <hip/hip_bf16.h>
#include <stddef.h>

// ---------- types ----------
typedef __attribute__((ext_vector_type(8)))  float  v8f;
typedef __attribute__((ext_vector_type(16))) __bf16 v16bf;

struct alignas(16) U4 { unsigned a, b, c, d; };
struct U8 { U4 lo; U4 hi; };

static __device__ __forceinline__ v16bf frag_cast(U4 lo, U4 hi) {
  U8 t; t.lo = lo; t.hi = hi;
  return __builtin_bit_cast(v16bf, t);
}

// one-VALU bf16 pair pack (truncation) via v_perm_b32:
// dst = [a.b2, a.b3, b.b2, b.b3]
static __device__ __forceinline__ unsigned pack2(float a, float b) {
  return __builtin_amdgcn_perm(__float_as_uint(b), __float_as_uint(a), 0x07060302u);
}
static __device__ __forceinline__ unsigned short f2bf(float a) {   // RNE
  unsigned ua = __float_as_uint(a);
  return (unsigned short)((ua + 0x7FFFu + ((ua >> 16) & 1u)) >> 16);
}
static __device__ __forceinline__ float bf2f(unsigned short h) {
  return __uint_as_float(((unsigned)h) << 16);
}

// ======================================================================
// Weight pre-pass: build BT_all[col][256] bf16, col-major-transposed.
// cols 0-255: W_off | 256-383: W_attn | 384-639: W_val | 640-895: W_out
// ======================================================================
__global__ __launch_bounds__(256)
void convert_wt(const float* __restrict__ W_off, const float* __restrict__ W_attn,
                const float* __restrict__ W_val, const float* __restrict__ W_out,
                unsigned short* __restrict__ BT) {
  int col = blockIdx.x, k = threadIdx.x;
  const float* src; int ldb, n;
  if      (col < 256) { src = W_off;  ldb = 256; n = col;       }
  else if (col < 384) { src = W_attn; ldb = 128; n = col - 256; }
  else if (col < 640) { src = W_val;  ldb = 256; n = col - 384; }
  else                { src = W_out;  ldb = 256; n = col - 640; }
  BT[(size_t)col * 256 + k] = f2bf(src[(size_t)k * ldb + n]);
}

// ======================================================================
// Tiled WMMA bf16 GEMM:  C[M,N] = A[M,256] * B[256,N] + bias  (+ epilogue)
// Block tile 64(M) x 128(N), 8 waves, each wave 32x32 via 2x2 WMMA frags,
// K-tile 64 = two WMMA k-steps per LDS fill (8 v_wmma per barrier pair).
// A: fp32 global -> v_perm bf16 pack -> LDS.  B: bf16 BT, fragments loaded
// straight from global with global_load_b128 pairs (contiguous k = 16h+s per lane half, as A).
// mode 0: bf16 -> vproj[(b*8+h)*nv + iv][32]   (value projection)
// mode 1: fp32 -> Cf[m*N + n]                  (qcat = off|attn)
// mode 2: fp32 Cout[m*N+n] = v + Resid[m*N+n]  (out proj + residual)
// ======================================================================
__global__ __launch_bounds__(256)
void wmma_gemm(const float* __restrict__ A, int M, int N,
               const unsigned short* __restrict__ BT, int bt_base,
               const float* __restrict__ bias0, const float* __restrict__ bias1,
               int nsplit, int mode,
               float* __restrict__ Cf,
               unsigned short* __restrict__ Cbf,
               float* __restrict__ Cout,
               const float* __restrict__ Resid,
               int nv)
{
  __shared__ unsigned As[64 * 32];    // [m][K/2] packed bf16 pairs, K-tile = 64

  const int tid  = threadIdx.x;
  const int lane = tid & 31;
  const int wid  = tid >> 5;
  const int wrow = wid >> 2;          // 0..1  (32 rows each)
  const int wcol = wid & 3;           // 0..3  (32 cols each)
  const int rowM0 = blockIdx.x * 64;
  const int colN0 = blockIdx.y * 128;

  v8f c[2][2];
#pragma unroll
  for (int i = 0; i < 2; i++)
#pragma unroll
    for (int j = 0; j < 2; j++)
      c[i][j] = (v8f)0.0f;

  for (int kk = 0; kk < 256; kk += 64) {
    // ---- A tile: 64x64 fp32 -> packed bf16 pairs in LDS (coalesced float4) ----
#pragma unroll
    for (int i = 0; i < 4; i++) {
      int p  = tid + i * 256;
      int m  = p >> 4, c4 = p & 15;           // c4*4 = starting column
      int row = rowM0 + m;
      if (row >= M) row = M - 1;              // clamp: no exec divergence
      const float4 v = *(const float4*)(A + (size_t)row * 256 + kk + c4 * 4);
      unsigned lo = pack2(v.x, v.y);
      unsigned hi = pack2(v.z, v.w);
      *(uint2*)&As[m * 32 + c4 * 2] = make_uint2(lo, hi);
      if (kk + 64 < 256)
        __builtin_prefetch(A + (size_t)row * 256 + kk + 64 + c4 * 4, 0, 3);
    }
    __syncthreads();

#pragma unroll
    for (int ks = 0; ks < 2; ks++) {          // two WMMA k-steps per LDS tile
      v16bf afrag[2], bfrag[2];
#pragma unroll
      for (int mi = 0; mi < 2; mi++) {
        int m = wrow * 32 + mi * 16 + (lane & 15);
        const U4* pb = (const U4*)&As[m * 32 + ks * 16 + 8 * (lane >> 4)];
        afrag[mi] = frag_cast(pb[0], pb[1]);  // K[base..base+15] contiguous
      }
#pragma unroll
      for (int nj = 0; nj < 2; nj++) {
        int n = bt_base + colN0 + wcol * 32 + nj * 16 + (lane & 15);
        const U4* pg = (const U4*)(BT + (size_t)n * 256 + kk + ks * 32 + 16 * (lane >> 4));
        bfrag[nj] = frag_cast(pg[0], pg[1]);  // K[base..base+15] contiguous
      }
#pragma unroll
      for (int mi = 0; mi < 2; mi++)
#pragma unroll
        for (int nj = 0; nj < 2; nj++)
          c[mi][nj] = __builtin_amdgcn_wmma_f32_16x16x32_bf16(
              false, afrag[mi], false, bfrag[nj], (short)0, c[mi][nj], false, false);
    }
    __syncthreads();
  }

  // ---- epilogue ----
#pragma unroll
  for (int mi = 0; mi < 2; mi++) {
#pragma unroll
    for (int nj = 0; nj < 2; nj++) {
      int nloc = wcol * 32 + nj * 16 + (lane & 15);
      int n    = colN0 + nloc;
      const float* biasp = (n >= nsplit) ? bias1 : bias0;
      int nb             = (n >= nsplit) ? (n - nsplit) : n;
      float bias = biasp[nb];
#pragma unroll
      for (int i = 0; i < 8; i++) {
        int m = rowM0 + wrow * 32 + mi * 16 + i + 8 * (lane >> 4);
        if (m < M) {
          float v = c[mi][nj][i] + bias;
          if (mode == 1) {
            Cf[(size_t)m * N + n] = v;
          } else if (mode == 2) {
            Cout[(size_t)m * N + n] = v + Resid[(size_t)m * N + n];
          } else {
            int b = m / nv, iv = m - b * nv;
            int h = n >> 5, ch = n & 31;
            Cbf[(((size_t)(b * 8 + h) * nv) + iv) * 32 + ch] = f2bf(v);
          }
        }
      }
    }
  }
}

// ======================================================================
// Per-(b,q) softmax + 16-point bilinear gather over bf16 vproj.
// 256 threads = 8 heads x 32 channels.
// ======================================================================
__global__ __launch_bounds__(256)
void msdeform_sample(const float* __restrict__ qcat,
                     const float* __restrict__ refp,
                     const unsigned short* __restrict__ vproj,
                     float* __restrict__ outs,
                     int nq, int nv)
{
  const int q = blockIdx.x, b = blockIdx.y;
  const int t = threadIdx.x;
  __shared__ float sX[128], sY[128], sW[128];

  const size_t qb   = (size_t)b * nq + q;
  const float* qrow = qcat + qb * 384;

  const int LHW[4] = {100, 50, 25, 13};
  const int LST[4] = {0, 10000, 12500, 13125};

  if (t < 128) {
    int h = t >> 4, j = t & 15, l = j >> 2;
    float Wf = (float)LHW[l], Hf = (float)LHW[l];   // square levels
    float rx = refp[(qb * 4 + l) * 2 + 0];
    float ry = refp[(qb * 4 + l) * 2 + 1];
    float ox = qrow[h * 32 + j * 2 + 0];
    float oy = qrow[h * 32 + j * 2 + 1];
    float lx = rx + ox * (1.0f / Wf);
    float ly = ry + oy * (1.0f / Hf);
    sX[t] = lx * Wf - 0.5f;                         // pixel-space x
    sY[t] = ly * Hf - 0.5f;                         // pixel-space y
    sW[t] = qrow[256 + h * 16 + j];                 // attn logit
  }
  __syncthreads();
  if (t < 8) {                                      // softmax over 16 pts per head
    float mx = -1e30f;
    for (int j = 0; j < 16; j++) mx = fmaxf(mx, sW[t * 16 + j]);
    float s = 0.f;
    for (int j = 0; j < 16; j++) { float e = __expf(sW[t * 16 + j] - mx); sW[t * 16 + j] = e; s += e; }
    float inv = 1.0f / s;
    for (int j = 0; j < 16; j++) sW[t * 16 + j] *= inv;
  }
  __syncthreads();

  const int h = t >> 5, ch = t & 31;
  const size_t vb = (size_t)(b * 8 + h) * nv * 32 + ch;
  float acc = 0.f;
#pragma unroll
  for (int j = 0; j < 16; j++) {
    int l = j >> 2;
    int W = LHW[l], H = LHW[l], S = LST[l];
    float px = sX[h * 16 + j], py = sY[h * 16 + j], w = sW[h * 16 + j];
    float fx0 = floorf(px), fy0 = floorf(py);
    int   x0 = (int)fx0, y0 = (int)fy0;
    int   x1 = x0 + 1,  y1 = y0 + 1;
    float ax = px - fx0, ay = py - fy0;
    float v00 = 0.f, v10 = 0.f, v01 = 0.f, v11 = 0.f;
    bool vx0 = (x0 >= 0) & (x0 < W), vx1 = (x1 >= 0) & (x1 < W);
    bool vy0 = (y0 >= 0) & (y0 < H), vy1 = (y1 >= 0) & (y1 < H);
    if (vx0 & vy0) v00 = bf2f(vproj[vb + (size_t)(S + y0 * W + x0) * 32]);
    if (vx1 & vy0) v10 = bf2f(vproj[vb + (size_t)(S + y0 * W + x1) * 32]);
    if (vx0 & vy1) v01 = bf2f(vproj[vb + (size_t)(S + y1 * W + x0) * 32]);
    if (vx1 & vy1) v11 = bf2f(vproj[vb + (size_t)(S + y1 * W + x1) * 32]);
    float top = v00 + ax * (v10 - v00);
    float bot = v01 + ax * (v11 - v01);
    acc += w * (top + ay * (bot - top));
  }
  outs[qb * 256 + t] = acc;   // t == h*32+ch
}

// ======================================================================
extern "C" void kernel_launch(void* const* d_in, const int* in_sizes, int n_in,
                              void* d_out, int out_size, void* d_ws, size_t ws_size,
                              hipStream_t stream) {
  const float* query  = (const float*)d_in[0];
  const float* value  = (const float*)d_in[1];
  const float* refp   = (const float*)d_in[2];
  const float* W_off  = (const float*)d_in[3];
  const float* b_off  = (const float*)d_in[4];
  const float* W_attn = (const float*)d_in[5];
  const float* b_attn = (const float*)d_in[6];
  const float* W_val  = (const float*)d_in[7];
  const float* b_val  = (const float*)d_in[8];
  const float* W_out  = (const float*)d_in[9];
  const float* b_out  = (const float*)d_in[10];

  const int NV = 13294;                       // 100^2+50^2+25^2+13^2
  int bs = in_sizes[1] / (NV * 256);
  int nq = in_sizes[0] / (bs * 256);
  int Mq = bs * nq;
  int Mv = bs * NV;

  // workspace layout
  char* ws = (char*)d_ws;
  size_t bt_bytes    = ((size_t)896 * 256 * 2 + 255) & ~(size_t)255;
  size_t qcat_bytes  = ((size_t)Mq * 384 * 4 + 255) & ~(size_t)255;
  size_t vproj_bytes = ((size_t)Mv * 256 * 2 + 255) & ~(size_t)255; // bs*8*nv*32 bf16
  unsigned short* BT    = (unsigned short*)ws;
  float*          qcat  = (float*)(ws + bt_bytes);
  unsigned short* vproj = (unsigned short*)(ws + bt_bytes + qcat_bytes);
  float*          outs  = (float*)(ws + bt_bytes + qcat_bytes + vproj_bytes);
  float*          out   = (float*)d_out;

  dim3 blk(256);

  // 0) transpose+convert all weights to bf16 BT layout
  convert_wt<<<dim3(896), blk, 0, stream>>>(W_off, W_attn, W_val, W_out, BT);

  // 1) qcat = query @ [W_off | W_attn] + [b_off | b_attn]   (N = 384, BT cols 0..383)
  wmma_gemm<<<dim3((Mq + 63) / 64, 3), blk, 0, stream>>>(
      query, Mq, 384, BT, 0, b_off, b_attn, 256,
      1, qcat, nullptr, nullptr, nullptr, NV);

  // 2) vproj = bf16( value @ W_val + b_val ) in [b*8+h][nv][32] layout (BT cols 384..639)
  wmma_gemm<<<dim3((Mv + 63) / 64, 2), blk, 0, stream>>>(
      value, Mv, 256, BT, 384, b_val, b_val, 1 << 30,
      0, nullptr, vproj, nullptr, nullptr, NV);

  // 3) softmax + bilinear sampling -> outs [bs*nq, 256]
  msdeform_sample<<<dim3(nq, bs), blk, 0, stream>>>(qcat, refp, vproj, outs, nq, NV);

  // 4) out = outs @ W_out + b_out + query   (BT cols 640..895)
  wmma_gemm<<<dim3((Mq + 63) / 64, 2), blk, 0, stream>>>(
      outs, Mq, 256, BT, 640, b_out, b_out, 1 << 30,
      2, nullptr, nullptr, out, query, NV);
}
